// FCWCRFLoss_28063316312223
// MI455X (gfx1250) — matrix operand order repaired and checked by reading, not compiled, not run on hardware
//
#include <hip/hip_runtime.h>
#include <hip/hip_bf16.h>

// ---------------------------------------------------------------------------
// FCWCRF loss on MI455X (gfx1250, wave32).
//   Stage 1 (prep):  per-pixel channel stats (var + 0.5/var so pairwise KL is
//                    FMA-only) + softmax(p) terms folded into
//                    u_k = sqrt(2)*q_k*E_k and q2_k = q_k^2 so the pairwise
//                    A*E*E term is mul+add+rcp+fma per class.
//                    Writes L2-normalized f in f16 (WMMA gram == cosine).
//   Stage 2 (pair):  128x128 (i,j) tiles; fn panels transposed into LDS;
//                    cos tile = one v_wmma_f32_16x16x32_f16 (K=32 == Cf);
//                    epilogue uses v_rcp_f32 (no IEEE div chains) + v_exp_f32.
//   Stage 3 (fin):   scalar finalize.
// ---------------------------------------------------------------------------

typedef __attribute__((ext_vector_type(16))) _Float16 v16h;
typedef __attribute__((ext_vector_type(8)))  _Float16 v8h;
typedef __attribute__((ext_vector_type(8)))  float    v8f;

#define B_    2
#define CF    32
#define NCLS  4
#define NPIX  2304          // 48*48
#define PP    3             // classes 1..3 after softmax slice
#define TILE  128
#define NT    (NPIX / TILE) // 18

__global__ void zero_kernel(float* acc) { *acc = 0.f; }

__global__ __launch_bounds__(256)
void prep_kernel(const float* __restrict__ f, const float* __restrict__ p,
                 float* __restrict__ mu, float* __restrict__ var,
                 float* __restrict__ rv,
                 float* __restrict__ q2, float* __restrict__ u,
                 _Float16* __restrict__ fn)
{
    int g = blockIdx.x * blockDim.x + threadIdx.x;
    if (g >= B_ * NPIX) return;
    int b = g / NPIX, n = g % NPIX;

    // channel stats (mean, ddof=1 variance, L2 norm)
    float fv[CF];
    float s = 0.f, ss = 0.f;
    const float* fb = f + (size_t)b * CF * NPIX + n;
    #pragma unroll
    for (int c = 0; c < CF; ++c) { float v = fb[c * NPIX]; fv[c] = v; s += v; ss += v * v; }
    float m   = s * (1.f / CF);
    float num = ss - CF * m * m; if (num < 0.f) num = 0.f;
    float v1  = num * (1.f / (CF - 1));
    mu[g]  = m;
    var[g] = v1;
    rv[g]  = 0.5f / v1;   // one divide per PIXEL instead of two per PAIR

    // normalized f16 copy: gram(fn) == cosine matrix
    float inv = rsqrtf(ss);
    _Float16* fnb = fn + (size_t)b * CF * NPIX + n;
    #pragma unroll
    for (int c = 0; c < CF; ++c) fnb[c * NPIX] = (_Float16)(fv[c] * inv);

    // softmax over 4 classes, keep 1..3
    const float* pb = p + (size_t)b * NCLS * NPIX + n;
    float pv[NCLS], mx = -1e30f;
    #pragma unroll
    for (int k = 0; k < NCLS; ++k) { pv[k] = pb[k * NPIX]; mx = fmaxf(mx, pv[k]); }
    float se = 0.f;
    #pragma unroll
    for (int k = 0; k < NCLS; ++k) { pv[k] = __expf(pv[k] - mx); se += pv[k]; }
    float rs = 1.f / se;
    const float kp  = 2.8853900817779268f; // CLASSES / ln(CLASSES)
    const float rt2 = 1.4142135623730951f; // sqrt(2)
    // Pairwise term A_k*E_ki*E_kj = (2 q_i q_j)/(q_i^2+q_j^2) * E_i * E_j
    //   = (u_ki * u_kj) / (q2_ki + q2_kj)  with u = sqrt(2)*q*E, q2 = q*q.
    #pragma unroll
    for (int k = 1; k < NCLS; ++k) {
        float qq = pv[k] * rs;
        float Ek = 1.f + kp * qq * __logf(qq);
        q2[(size_t)(k - 1) * (B_ * NPIX) + g] = qq * qq;
        u [(size_t)(k - 1) * (B_ * NPIX) + g] = rt2 * qq * Ek;
    }
}

__global__ __launch_bounds__(256)
void pair_kernel(const _Float16* __restrict__ fn,
                 const float* __restrict__ mu, const float* __restrict__ var,
                 const float* __restrict__ rv,
                 const float* __restrict__ q2, const float* __restrict__ u,
                 float* __restrict__ acc)
{
    // fn panels stored [pixel][channel] so each lane's WMMA fragment is two
    // contiguous 16B runs -> ds_read_b128 pairs.
    __shared__ __attribute__((aligned(16))) _Float16 sfnI[TILE][CF];
    __shared__ __attribute__((aligned(16))) _Float16 sfnJ[TILE][CF];
    __shared__ float sMuI[TILE], sVarI[TILE], sRvI[TILE];
    __shared__ float sMuJ[TILE], sVarJ[TILE], sRvJ[TILE];
    __shared__ float sQ2I[PP][TILE], sUI[PP][TILE], sQ2J[PP][TILE], sUJ[PP][TILE];
    __shared__ float sred[8];

    const int tid = threadIdx.x;
    const int b  = blockIdx.z;
    const int i0 = blockIdx.y * TILE;
    const int j0 = blockIdx.x * TILE;
    const size_t pixbase = (size_t)b * NPIX;
    const _Float16* fnb = fn + (size_t)b * CF * NPIX;

    // stage transposed fn panels (coalesced global reads along n)
    for (int idx = tid; idx < CF * TILE; idx += 256) {
        int k  = idx >> 7;          // 0..31
        int nl = idx & (TILE - 1);  // 0..127
        sfnI[nl][k] = fnb[(size_t)k * NPIX + i0 + nl];
        sfnJ[nl][k] = fnb[(size_t)k * NPIX + j0 + nl];
    }
    if (tid < TILE) {
        sMuI[tid]  = mu[pixbase + i0 + tid];
        sVarI[tid] = var[pixbase + i0 + tid];
        sRvI[tid]  = rv[pixbase + i0 + tid];
        sMuJ[tid]  = mu[pixbase + j0 + tid];
        sVarJ[tid] = var[pixbase + j0 + tid];
        sRvJ[tid]  = rv[pixbase + j0 + tid];
        #pragma unroll
        for (int k = 0; k < PP; ++k) {
            size_t o = (size_t)k * (B_ * NPIX) + pixbase;
            sQ2I[k][tid] = q2[o + i0 + tid];
            sUI [k][tid] = u [o + i0 + tid];
            sQ2J[k][tid] = q2[o + j0 + tid];
            sUJ [k][tid] = u [o + j0 + tid];
        }
    }
    __syncthreads();

    const int wv   = tid >> 5;   // wave 0..7 -> i-subtile row
    const int lane = tid & 31;
    const int h    = lane >> 4;  // half-wave
    const int lm   = lane & 15;

    // A fragment (16x32 f16): lane half h holds K in [8h,8h+8) then [16+8h,...)
    const int irow = wv * 16 + lm;
    v8h alo = *(const v8h*)&sfnI[irow][h * 8];
    v8h ahi = *(const v8h*)&sfnI[irow][16 + h * 8];
    v16h afrag;
    #pragma unroll
    for (int e = 0; e < 8; ++e) { afrag[e] = alo[e]; afrag[8 + e] = ahi[e]; }

    float accT = 0.f;
    #pragma unroll
    for (int sj = 0; sj < 8; ++sj) {
        // B fragment (32x16 f16): lane half h holds K in [8h,8h+8) then [16+8h,...)
        const int jcol = sj * 16 + lm;
        v8h blo = *(const v8h*)&sfnJ[jcol][h * 8];
        v8h bhi = *(const v8h*)&sfnJ[jcol][16 + h * 8];
        v16h bfrag;
        #pragma unroll
        for (int e = 0; e < 8; ++e) { bfrag[e] = blo[e]; bfrag[8 + e] = bhi[e]; }

        v8f c = {};
        c = __builtin_amdgcn_wmma_f32_16x16x32_f16(false, afrag, false, bfrag,
                                                   (short)0, c, false, false);

        const float muj = sMuJ[jcol], vj = sVarJ[jcol], rvj = sRvJ[jcol];
        const float qj0 = sQ2J[0][jcol], qj1 = sQ2J[1][jcol], qj2 = sQ2J[2][jcol];
        const float uj0 = sUJ [0][jcol], uj1 = sUJ [1][jcol], uj2 = sUJ [2][jcol];

        // C/D layout: VGPR r -> M = r + 8*(lane>=16); N = lane%16
        #pragma unroll
        for (int r = 0; r < 8; ++r) {
            const int il = wv * 16 + h * 8 + r;
            float mui = sMuI[il], vi = sVarI[il], rvi = sRvI[il];
            float d  = mui - muj;
            float d2 = d * d;
            // dist = (KL + 2(1-cos))/3, KL = 0.5*(t1+t2-1), log-ratio == 0:
            //   uKL = t1 + t2 - 4*cos + 3 = 6*dist; -4*dist^2 = -uKL^2/9
            float t1  = (vi + d2) * rvj;
            float t2  = (vj + d2) * rvi;
            float uKL = t1 + t2 - 4.f * c[r] + 3.f;
            float w   = __expf(uKL * uKL * (-1.f / 9.f));
            // S = sum_k (u_ki*u_kj) * rcp(q2_ki + q2_kj)  [single v_rcp_f32 each]
            float S = (sUI[0][il] * uj0) * __builtin_amdgcn_rcpf(sQ2I[0][il] + qj0)
                    + (sUI[1][il] * uj1) * __builtin_amdgcn_rcpf(sQ2I[1][il] + qj1)
                    + (sUI[2][il] * uj2) * __builtin_amdgcn_rcpf(sQ2I[2][il] + qj2);
            accT = fmaf(w, S, accT);
        }
    }

    // wave32 butterfly reduce, then one atomic per block
    #pragma unroll
    for (int off = 16; off > 0; off >>= 1)
        accT += __shfl_xor(accT, off, 32);
    if (lane == 0) sred[wv] = accT;
    __syncthreads();
    if (tid == 0) {
        float t = 0.f;
        #pragma unroll
        for (int wvi = 0; wvi < 8; ++wvi) t += sred[wvi];
        atomicAdd(acc, t);
    }
}

__global__ void finalize_kernel(const float* __restrict__ acc, float* __restrict__ out)
{
    float total = *acc;
    float score = total / (float)(B_ * PP * NPIX);
    out[0] = 1.f - score / (float)NPIX;
}

extern "C" void kernel_launch(void* const* d_in, const int* in_sizes, int n_in,
                              void* d_out, int out_size, void* d_ws, size_t ws_size,
                              hipStream_t stream)
{
    const float* f = (const float*)d_in[0];  // (2,32,48,48)
    const float* p = (const float*)d_in[1];  // (2,4,48,48)
    float* out = (float*)d_out;

    char* w = (char*)d_ws;
    size_t off = 0;
    float* acc = (float*)(w + off); off += 256;                      // scalar accumulator
    const size_t BN = (size_t)B_ * NPIX;
    float* mu  = (float*)(w + off); off += BN * sizeof(float);
    float* var = (float*)(w + off); off += BN * sizeof(float);
    float* rv  = (float*)(w + off); off += BN * sizeof(float);
    float* q2  = (float*)(w + off); off += (size_t)PP * BN * sizeof(float);
    float* u   = (float*)(w + off); off += (size_t)PP * BN * sizeof(float);
    _Float16* fn = (_Float16*)(w + off); off += (size_t)B_ * CF * NPIX * sizeof(_Float16);

    zero_kernel<<<1, 1, 0, stream>>>(acc);
    prep_kernel<<<(B_ * NPIX + 255) / 256, 256, 0, stream>>>(f, p, mu, var, rv, q2, u, fn);
    dim3 grid(NT, NT, B_);
    pair_kernel<<<grid, 256, 0, stream>>>(fn, mu, var, rv, q2, u, acc);
    finalize_kernel<<<1, 1, 0, stream>>>(acc, out);
}
